// BipartiteGCNNodeSelectionPolicy_57827439674233
// MI455X (gfx1250) — matrix operand order repaired and checked by reading, not compiled, not run on hardware
//
#include <hip/hip_runtime.h>

// ---------------------------------------------------------------------------
// CDNA5 (gfx1250) bf16 WMMA types
// ---------------------------------------------------------------------------
typedef __bf16 v16bf __attribute__((ext_vector_type(16)));
typedef __bf16 v8bf  __attribute__((ext_vector_type(8)));
typedef float  v8f   __attribute__((ext_vector_type(8)));

// Manual RNE f32->bf16 (used in non-hot prep/init kernels)
__device__ __forceinline__ unsigned short f2bf(float f) {
  unsigned int u = __float_as_uint(f);
  return (unsigned short)((u + 0x7FFFu + ((u >> 16) & 1u)) >> 16);
}

// Native f32->bf16 (hot path; hardware cvt, round-to-nearest-even)
__device__ __forceinline__ unsigned short f2bf_n(float f) {
  union { __bf16 h; unsigned short u; } cv;
  cv.h = (__bf16)f;
  return cv.u;
}

__device__ __forceinline__ v16bf cat16(v8bf lo, v8bf hi) {
  return __builtin_shufflevector(lo, hi, 0, 1, 2, 3, 4, 5, 6, 7,
                                         8, 9, 10, 11, 12, 13, 14, 15);
}

// A-matrix fragment (16x32 bf16) per ISA layout:
// lane&15 = row M; (lane>>4) selects K-half; elements 0..7 -> K=kb..kb+7,
// elements 8..15 -> K=kb+16..kb+23.  A is an LDS tile [16][KA] bf16.
__device__ __forceinline__ v16bf load_a_frag(const unsigned short* A, int KA,
                                             int kt, int lane) {
  const unsigned short* p = A + (lane & 15) * KA + kt * 32 + ((lane >> 4) << 3);
  v8bf lo = *(const v8bf*)p;
  v8bf hi = *(const v8bf*)(p + 16);
  return cat16(lo, hi);
}

// B fragments are pre-swizzled so each lane reads 16 contiguous bf16:
// flat index ((kt*4+nt)*32 + lane)*16 + e
//   value = W[kt*32 + (e>>3)*16 + (lane>>4)*8 + (e&7)][nt*16 + (lane&15)]
__device__ __forceinline__ v16bf load_b_frag(const unsigned short* F, int ktnt,
                                             int lane) {
  const unsigned short* p = F + (((unsigned)ktnt << 5) + (unsigned)lane) * 16u;
  v8bf lo = *(const v8bf*)p;
  v8bf hi = *(const v8bf*)(p + 8);
  return cat16(lo, hi);
}

__device__ __forceinline__ v8f wmma_bf16(v16bf a, v16bf b, v8f c) {
  return __builtin_amdgcn_wmma_f32_16x16x32_bf16(false, a, false, b, (short)0,
                                                 c, false, false);
}

// ---------------------------------------------------------------------------
// Weight prep: fp32 [K,64] row-major -> bf16 B-fragment layout (zero-padded K)
// ---------------------------------------------------------------------------
__global__ void prep_frags_kernel(const float* __restrict__ W, int Kreal,
                                  int KT, unsigned short* __restrict__ out) {
  int n = KT * 4 * 512;
  for (int i = blockIdx.x * blockDim.x + threadIdx.x; i < n;
       i += gridDim.x * blockDim.x) {
    int e = i & 15;
    int lane = (i >> 4) & 31;
    int nt = (i >> 9) & 3;
    int kt = i >> 11;
    int K = kt * 32 + ((e < 8) ? ((lane >> 4) * 8 + e) : (16 + (lane >> 4) * 8 + (e - 8)));
    int c = nt * 16 + (lane & 15);
    float v = (K < Kreal) ? W[K * 64 + c] : 0.f;
    out[i] = f2bf(v);
  }
}

__global__ void zero_kernel(float* __restrict__ p, long long n) {
  for (long long i = blockIdx.x * (long long)blockDim.x + threadIdx.x; i < n;
       i += (long long)gridDim.x * blockDim.x)
    p[i] = 0.f;
}

// ---------------------------------------------------------------------------
// Node init: emb = relu(x @ W + b), stored bf16
// ---------------------------------------------------------------------------
__global__ void node_init_kernel(const float* __restrict__ xv,
                                 const float* __restrict__ xc,
                                 const float* __restrict__ vW,
                                 const float* __restrict__ vb,
                                 const float* __restrict__ cW,
                                 const float* __restrict__ cb,
                                 unsigned short* __restrict__ v0h,
                                 unsigned short* __restrict__ c0h, int Nv,
                                 int Nc) {
  int gid = blockIdx.x * blockDim.x + threadIdx.x;
  int nv = Nv * 64;
  if (gid < nv) {
    int n = gid >> 6, j = gid & 63;
    float a = vb[j];
#pragma unroll
    for (int k = 0; k < 9; k++) a += xv[n * 9 + k] * vW[k * 64 + j];
    v0h[gid] = f2bf(fmaxf(a, 0.f));
  } else if (gid < nv + Nc * 64) {
    int g = gid - nv;
    int n = g >> 6, j = g & 63;
    float a = cb[j];
#pragma unroll
    for (int k = 0; k < 5; k++) a += xc[n * 5 + k] * cW[k * 64 + j];
    c0h[g] = f2bf(fmaxf(a, 0.f));
  }
}

// ---------------------------------------------------------------------------
// Edge message kernel: msg = relu([src|dst|ea] @ g1 + g1b) @ g2 + g2b
// then atomicAdd into agg[dst].  One 16-edge tile per wave iteration.
// All 28 weight B-fragments (g1: 5x4, g2: 2x4) stay resident in VGPRs.
// ---------------------------------------------------------------------------
#define EK_WAVES 4
__global__ __launch_bounds__(128) void edge_msg_kernel(
    const unsigned short* __restrict__ src_bf,
    const unsigned short* __restrict__ dst_bf,
    const int* __restrict__ src_idx, const int* __restrict__ dst_idx,
    const float* __restrict__ ea, const unsigned short* __restrict__ g1_frags,
    const unsigned short* __restrict__ g2_frags,
    const float* __restrict__ g1b, const float* __restrict__ g2b,
    float* __restrict__ agg, int E) {
  __shared__ __align__(16) unsigned short sm_a[EK_WAVES * 16 * 160];  // 20 KB
  __shared__ __align__(16) unsigned short sm_h[EK_WAVES * 16 * 64];   // 8 KB
  int tid = threadIdx.x;
  int wave = tid >> 5, lane = tid & 31;
  unsigned short* A = sm_a + wave * 16 * 160;
  unsigned short* H = sm_h + wave * 16 * 64;
  int col = lane & 15;
  int rbase = (lane >> 4) << 3;

  // hoist loop-invariant weight fragments into registers (224 VGPRs)
  v16bf bg1[20];
#pragma unroll
  for (int i = 0; i < 20; i++) bg1[i] = load_b_frag(g1_frags, i, lane);
  v16bf bg2[8];
#pragma unroll
  for (int i = 0; i < 8; i++) bg2[i] = load_b_frag(g2_frags, i, lane);
  float bias1[4], bias2[4];
#pragma unroll
  for (int nt = 0; nt < 4; nt++) {
    bias1[nt] = g1b[nt * 16 + col];
    bias2[nt] = g2b[nt * 16 + col];
  }

  int ntile = (E + 15) >> 4;
  for (int tile = blockIdx.x * EK_WAVES + wave; tile < ntile;
       tile += gridDim.x * EK_WAVES) {
    // ---- stage A tile: 16 rows x [src(64)|dst(64)|ea|pad31] bf16 ----
#pragma unroll 1
    for (int r = 0; r < 16; ++r) {
      int e = tile * 16 + r;
      if (e >= E) e = E - 1;
      int si = src_idx[e], di = dst_idx[e];
      const unsigned int* srow = (const unsigned int*)(src_bf + (size_t)si * 64);
      const unsigned int* drow = (const unsigned int*)(dst_bf + (size_t)di * 64);
      unsigned int* arow = (unsigned int*)(A + r * 160);
      arow[lane] = srow[lane];
      arow[32 + lane] = drow[lane];
      if (lane < 16)
        arow[64 + lane] = (lane == 0) ? (unsigned int)f2bf_n(ea[e]) : 0u;
    }
    // ---- g1: [16x160] @ [160x64], 5 k-steps x 4 n-tiles ----
    v16bf af[5];
#pragma unroll
    for (int kt = 0; kt < 5; kt++) af[kt] = load_a_frag(A, 160, kt, lane);
#pragma unroll
    for (int nt = 0; nt < 4; nt++) {
      v8f acc = {0.f, 0.f, 0.f, 0.f, 0.f, 0.f, 0.f, 0.f};
#pragma unroll
      for (int kt = 0; kt < 5; kt++)
        acc = wmma_bf16(af[kt], bg1[kt * 4 + nt], acc);
      float b = bias1[nt];
#pragma unroll
      for (int i = 0; i < 8; i++) {
        float x = fmaxf(acc[i] + b, 0.f);
        H[(rbase + i) * 64 + nt * 16 + col] = f2bf_n(x);
      }
    }
    // ---- g2: [16x64] @ [64x64] ----
    v16bf hf[2];
#pragma unroll
    for (int kt = 0; kt < 2; kt++) hf[kt] = load_a_frag(H, 64, kt, lane);
#pragma unroll
    for (int nt = 0; nt < 4; nt++) {
      v8f acc = {0.f, 0.f, 0.f, 0.f, 0.f, 0.f, 0.f, 0.f};
#pragma unroll
      for (int kt = 0; kt < 2; kt++)
        acc = wmma_bf16(hf[kt], bg2[kt * 4 + nt], acc);
      float b = bias2[nt];
#pragma unroll
      for (int i = 0; i < 8; i++) {
        int e = tile * 16 + rbase + i;
        if (e < E) {
          int di = dst_idx[e];
          atomicAdd(&agg[(size_t)di * 64 + nt * 16 + col], acc[i] + b);
        }
      }
    }
  }
}

// ---------------------------------------------------------------------------
// Column stats: per-column sum / sumsq over [N,64] (for BN and mean-pool)
// ---------------------------------------------------------------------------
__global__ void colstats_kernel(const float* __restrict__ x, int N,
                                float* __restrict__ sums,
                                float* __restrict__ sumsq) {
  int c = threadIdx.x & 63, rg = threadIdx.x >> 6;  // blockDim = 256
  float s = 0.f, q = 0.f;
  for (int r = blockIdx.x * 4 + rg; r < N; r += gridDim.x * 4) {
    float v = x[(size_t)r * 64 + c];
    s += v;
    q += v * v;
  }
  __shared__ float sh[512];
  sh[threadIdx.x] = s;
  sh[256 + threadIdx.x] = q;
  __syncthreads();
  if (rg == 0) {
#pragma unroll
    for (int i = 1; i < 4; i++) {
      s += sh[i * 64 + c];
      q += sh[256 + i * 64 + c];
    }
    atomicAdd(&sums[c], s);
    atomicAdd(&sumsq[c], q);
  }
}

// ---------------------------------------------------------------------------
// Node update: out = relu([dst | BN(agg)] @ f1 + f1b) @ f2 + f2b
// Weight B-fragments (f1: 4x4, f2: 2x4) resident in VGPRs.
// ---------------------------------------------------------------------------
__global__ __launch_bounds__(128) void node_update_kernel(
    const unsigned short* __restrict__ dst_bf, const float* __restrict__ agg,
    const float* __restrict__ sums, const float* __restrict__ sumsq,
    const float* __restrict__ bn_g, const float* __restrict__ bn_b,
    const unsigned short* __restrict__ f1_frags,
    const unsigned short* __restrict__ f2_frags,
    const float* __restrict__ f1b, const float* __restrict__ f2b,
    float* __restrict__ out_f, unsigned short* __restrict__ out_h, int Nd) {
  __shared__ __align__(16) unsigned short sm_a[4 * 16 * 128];  // 16 KB
  __shared__ __align__(16) unsigned short sm_h[4 * 16 * 64];   // 8 KB
  int tid = threadIdx.x;
  int wave = tid >> 5, lane = tid & 31;
  unsigned short* A = sm_a + wave * 16 * 128;
  unsigned short* H = sm_h + wave * 16 * 64;
  int col = lane & 15;
  int rbase = (lane >> 4) << 3;

  v16bf bf1[16];
#pragma unroll
  for (int i = 0; i < 16; i++) bf1[i] = load_b_frag(f1_frags, i, lane);
  v16bf bf2[8];
#pragma unroll
  for (int i = 0; i < 8; i++) bf2[i] = load_b_frag(f2_frags, i, lane);
  float bias1[4], bias2[4];
#pragma unroll
  for (int nt = 0; nt < 4; nt++) {
    bias1[nt] = f1b[nt * 16 + col];
    bias2[nt] = f2b[nt * 16 + col];
  }

  // BN affine folded: aggn = agg*sc + sh ; this lane owns columns 2L, 2L+1
  int c0 = lane * 2, c1 = lane * 2 + 1;
  float inv = 1.f / (float)Nd;
  float m0 = sums[c0] * inv, m1 = sums[c1] * inv;
  float va0 = sumsq[c0] * inv - m0 * m0, va1 = sumsq[c1] * inv - m1 * m1;
  float sc0 = rsqrtf(va0 + 1e-5f) * bn_g[c0];
  float sc1 = rsqrtf(va1 + 1e-5f) * bn_g[c1];
  float sh0 = bn_b[c0] - m0 * sc0, sh1 = bn_b[c1] - m1 * sc1;

  int ntile = (Nd + 15) >> 4;
  for (int tile = blockIdx.x * 4 + wave; tile < ntile; tile += gridDim.x * 4) {
#pragma unroll 1
    for (int r = 0; r < 16; ++r) {
      int n = tile * 16 + r;
      if (n >= Nd) n = Nd - 1;
      const unsigned int* drow = (const unsigned int*)(dst_bf + (size_t)n * 64);
      unsigned int* arow = (unsigned int*)(A + r * 128);
      arow[lane] = drow[lane];
      const float2* grow = (const float2*)(agg + (size_t)n * 64);
      float2 g = grow[lane];
      float a0 = g.x * sc0 + sh0;
      float a1 = g.y * sc1 + sh1;
      arow[32 + lane] =
          (unsigned int)f2bf_n(a0) | ((unsigned int)f2bf_n(a1) << 16);
    }
    v16bf af[4];
#pragma unroll
    for (int kt = 0; kt < 4; kt++) af[kt] = load_a_frag(A, 128, kt, lane);
#pragma unroll
    for (int nt = 0; nt < 4; nt++) {
      v8f acc = {0.f, 0.f, 0.f, 0.f, 0.f, 0.f, 0.f, 0.f};
#pragma unroll
      for (int kt = 0; kt < 4; kt++)
        acc = wmma_bf16(af[kt], bf1[kt * 4 + nt], acc);
      float b = bias1[nt];
#pragma unroll
      for (int i = 0; i < 8; i++) {
        float x = fmaxf(acc[i] + b, 0.f);
        H[(rbase + i) * 64 + nt * 16 + col] = f2bf_n(x);
      }
    }
    v16bf hf[2];
#pragma unroll
    for (int kt = 0; kt < 2; kt++) hf[kt] = load_a_frag(H, 64, kt, lane);
#pragma unroll
    for (int nt = 0; nt < 4; nt++) {
      v8f acc = {0.f, 0.f, 0.f, 0.f, 0.f, 0.f, 0.f, 0.f};
#pragma unroll
      for (int kt = 0; kt < 2; kt++)
        acc = wmma_bf16(hf[kt], bf2[kt * 4 + nt], acc);
      float b = bias2[nt];
#pragma unroll
      for (int i = 0; i < 8; i++) {
        int n = tile * 16 + rbase + i;
        if (n < Nd) {
          float v = acc[i] + b;
          out_f[(size_t)n * 64 + nt * 16 + col] = v;
          out_h[(size_t)n * 64 + nt * 16 + col] = f2bf_n(v);
        }
      }
    }
  }
}

// ---------------------------------------------------------------------------
// Head: score = relu([pooled | s] @ W1 + b1) @ W2 + b2
// ---------------------------------------------------------------------------
__global__ void head_kernel(const float* __restrict__ pooled_sums, int Nv,
                            const float* __restrict__ cand,
                            const float* __restrict__ W1,
                            const float* __restrict__ b1,
                            const float* __restrict__ W2,
                            const float* __restrict__ b2,
                            float* __restrict__ out) {
  __shared__ float x[66];
  __shared__ float h[64];
  int t = threadIdx.x;  // 64 threads
  x[t] = pooled_sums[t] / (float)Nv;
  if (t < 2) {
    float cs = cand[t];
    x[64 + t] = cs / (fabsf(cs) + 1.f);
  }
  __syncthreads();
  float a = b1[t];
#pragma unroll 1
  for (int k = 0; k < 66; k++) a += x[k] * W1[k * 64 + t];
  h[t] = fmaxf(a, 0.f) * W2[t];
  __syncthreads();
  if (t == 0) {
    float s = b2[0];
    for (int k = 0; k < 64; k++) s += h[k];
    out[0] = s;
  }
}

// ---------------------------------------------------------------------------
// Launch
// ---------------------------------------------------------------------------
extern "C" void kernel_launch(void* const* d_in, const int* in_sizes, int n_in,
                              void* d_out, int out_size, void* d_ws,
                              size_t ws_size, hipStream_t stream) {
  (void)n_in; (void)out_size; (void)ws_size;
  const float* x_var = (const float*)d_in[0];
  const float* x_con = (const float*)d_in[1];
  const int* eidx = (const int*)d_in[2];
  const float* eattr = (const float*)d_in[3];
  const float* cand = (const float*)d_in[4];
  // JAX pytree flattening: dict keys sorted alphabetically.
  // params: c_to_v{bn_b,bn_g,f1W,f1b,f2W,f2b,g1W,g1b,g2W,g2b}, con_init{W,b},
  //         head{W1,W2,b1,b2}, v_to_c{...}, var_init{W,b}
  const float* ctv_bnb = (const float*)d_in[5];
  const float* ctv_bng = (const float*)d_in[6];
  const float* ctv_f1W = (const float*)d_in[7];
  const float* ctv_f1b = (const float*)d_in[8];
  const float* ctv_f2W = (const float*)d_in[9];
  const float* ctv_f2b = (const float*)d_in[10];
  const float* ctv_g1W = (const float*)d_in[11];
  const float* ctv_g1b = (const float*)d_in[12];
  const float* ctv_g2W = (const float*)d_in[13];
  const float* ctv_g2b = (const float*)d_in[14];
  const float* con_W = (const float*)d_in[15];
  const float* con_b = (const float*)d_in[16];
  const float* head_W1 = (const float*)d_in[17];
  const float* head_W2 = (const float*)d_in[18];
  const float* head_b1 = (const float*)d_in[19];
  const float* head_b2 = (const float*)d_in[20];
  const float* vtc_bnb = (const float*)d_in[21];
  const float* vtc_bng = (const float*)d_in[22];
  const float* vtc_f1W = (const float*)d_in[23];
  const float* vtc_f1b = (const float*)d_in[24];
  const float* vtc_f2W = (const float*)d_in[25];
  const float* vtc_f2b = (const float*)d_in[26];
  const float* vtc_g1W = (const float*)d_in[27];
  const float* vtc_g1b = (const float*)d_in[28];
  const float* vtc_g2W = (const float*)d_in[29];
  const float* vtc_g2b = (const float*)d_in[30];
  const float* var_W = (const float*)d_in[31];
  const float* var_b = (const float*)d_in[32];

  int Nv = in_sizes[0] / 9;
  int Nc = in_sizes[1] / 5;
  int E = in_sizes[3];  // edge_attr has EDGE_DIM=1
  const int* con_idx = eidx;      // edge_index[0]
  const int* var_idx = eidx + E;  // edge_index[1]

  // ---- carve workspace ----
  char* ws = (char*)d_ws;
  size_t off = 0;
  auto carve = [&](size_t bytes) -> char* {
    char* p = ws + off;
    off = (off + bytes + 255) & ~(size_t)255;
    return p;
  };
  unsigned short* v0h = (unsigned short*)carve((size_t)Nv * 64 * 2);
  unsigned short* c0h = (unsigned short*)carve((size_t)Nc * 64 * 2);
  unsigned short* c1h = (unsigned short*)carve((size_t)Nc * 64 * 2);
  unsigned short* v1h = (unsigned short*)carve((size_t)Nv * 64 * 2);
  float* c1f = (float*)carve((size_t)Nc * 64 * 4);
  float* v1f = (float*)carve((size_t)Nv * 64 * 4);
  float* aggc = (float*)carve((size_t)Nc * 64 * 4);
  float* aggv = (float*)carve((size_t)Nv * 64 * 4);
  float* stats = (float*)carve(6 * 64 * 4);
  float* sums_c = stats, *sumsq_c = stats + 64;
  float* sums_v = stats + 128, *sumsq_v = stats + 192;
  float* pool_s = stats + 256, *pool_q = stats + 320;
  unsigned short* vtc_g1f = (unsigned short*)carve(5 * 4 * 512 * 2);
  unsigned short* vtc_g2f = (unsigned short*)carve(2 * 4 * 512 * 2);
  unsigned short* vtc_f1f = (unsigned short*)carve(4 * 4 * 512 * 2);
  unsigned short* vtc_f2f = (unsigned short*)carve(2 * 4 * 512 * 2);
  unsigned short* ctv_g1f = (unsigned short*)carve(5 * 4 * 512 * 2);
  unsigned short* ctv_g2f = (unsigned short*)carve(2 * 4 * 512 * 2);
  unsigned short* ctv_f1f = (unsigned short*)carve(4 * 4 * 512 * 2);
  unsigned short* ctv_f2f = (unsigned short*)carve(2 * 4 * 512 * 2);

  // ---- weight prep (bf16 B-fragment swizzle) ----
  auto prep = [&](const float* W, int Kreal, int KT, unsigned short* o) {
    int n = KT * 4 * 512;
    prep_frags_kernel<<<(n + 255) / 256, 256, 0, stream>>>(W, Kreal, KT, o);
  };
  prep(vtc_g1W, 129, 5, vtc_g1f);
  prep(vtc_g2W, 64, 2, vtc_g2f);
  prep(vtc_f1W, 128, 4, vtc_f1f);
  prep(vtc_f2W, 64, 2, vtc_f2f);
  prep(ctv_g1W, 129, 5, ctv_g1f);
  prep(ctv_g2W, 64, 2, ctv_g2f);
  prep(ctv_f1W, 128, 4, ctv_f1f);
  prep(ctv_f2W, 64, 2, ctv_f2f);

  // ---- zero accumulators (aggc, aggv, stats are contiguous in ws) ----
  long long zn = (long long)(((char*)(stats + 6 * 64)) - (char*)aggc) / 4;
  zero_kernel<<<2048, 256, 0, stream>>>(aggc, zn);

  // ---- node init ----
  int tinit = (Nv + Nc) * 64;
  node_init_kernel<<<(tinit + 255) / 256, 256, 0, stream>>>(
      x_var, x_con, var_W, var_b, con_W, con_b, v0h, c0h, Nv, Nc);

  // ---- v -> c ----
  edge_msg_kernel<<<2048, 128, 0, stream>>>(v0h, c0h, var_idx, con_idx, eattr,
                                            vtc_g1f, vtc_g2f, vtc_g1b, vtc_g2b,
                                            aggc, E);
  colstats_kernel<<<512, 256, 0, stream>>>(aggc, Nc, sums_c, sumsq_c);
  node_update_kernel<<<512, 128, 0, stream>>>(
      c0h, aggc, sums_c, sumsq_c, vtc_bng, vtc_bnb, vtc_f1f, vtc_f2f, vtc_f1b,
      vtc_f2b, c1f, c1h, Nc);

  // ---- c -> v ----
  edge_msg_kernel<<<2048, 128, 0, stream>>>(c1h, v0h, con_idx, var_idx, eattr,
                                            ctv_g1f, ctv_g2f, ctv_g1b, ctv_g2b,
                                            aggv, E);
  colstats_kernel<<<512, 256, 0, stream>>>(aggv, Nv, sums_v, sumsq_v);
  node_update_kernel<<<1024, 128, 0, stream>>>(
      v0h, aggv, sums_v, sumsq_v, ctv_bng, ctv_bnb, ctv_f1f, ctv_f2f, ctv_f1b,
      ctv_f2b, v1f, v1h, Nv);

  // ---- mean pool + head ----
  colstats_kernel<<<512, 256, 0, stream>>>(v1f, Nv, pool_s, pool_q);
  head_kernel<<<1, 64, 0, stream>>>(pool_s, Nv, cand, head_W1, head_b1,
                                    head_W2, head_b2, (float*)d_out);
}
